// AttentionCell_23983097380925
// MI455X (gfx1250) — matrix operand order repaired and checked by reading, not compiled, not run on hardware
//
#include <hip/hip_runtime.h>
#include <hip/hip_bf16.h>
#include <cstdint>

// ---------------------------------------------------------------------------
// Problem constants (from reference): B=512, T=256, C=512, H=512, NC=96
// ---------------------------------------------------------------------------
#define B_   512
#define T_   256
#define C_   512
#define H_   512
#define NC_  96
#define H4_  2048          // 4*H
#define KEFF 1120          // (C + NC) + H = 608 + 512  (LSTM reduction depth)
#define LDA  528           // padded LDS row stride in bf16 elems (1056 B, 16B-aligned)

typedef __attribute__((ext_vector_type(16))) __bf16 v16bf;
typedef __attribute__((ext_vector_type(8)))  float  v8f;
typedef __attribute__((ext_vector_type(2)))  float  v2f;

union BFV { v16bf v; uint4 q[2]; };   // 32 bytes: one 16-half WMMA operand

// fp32 -> bf16 round-to-nearest-even (bit pattern)
static __device__ __forceinline__ unsigned short f2bf(float f) {
    unsigned int u = __float_as_uint(f);
    unsigned int r = u + 0x7FFFu + ((u >> 16) & 1u);
    return (unsigned short)(r >> 16);
}

static __device__ __forceinline__ float sigmoidf(float x) {
    return 1.0f / (1.0f + expf(-x));
}

// ---------------------------------------------------------------------------
// Kernel H: hp = prev_h @ Wh + bh   ([512,512]x[512,512], fp32 WMMA).
// Hoisted out of the attention kernel so Wh is streamed from L2 ONCE
// (previously every attention WG re-read the full 1 MB Wh -> 512 MB L2).
// Grid (H/32, B/64); 256 threads = 8 waves (4 M x 2 N), 16x16 frag per wave.
// ---------------------------------------------------------------------------
__global__ __launch_bounds__(256)
void hp_kernel(const float* __restrict__ prev_h,   // [B, H]
               const float* __restrict__ Wh,       // [H, H]
               const float* __restrict__ bh,       // [H]
               float* __restrict__ hp)             // [B, H] workspace
{
    __shared__ float As[64][33];
    __shared__ float Bs[32][33];

    const int h0 = blockIdx.x * 32;
    const int m0 = blockIdx.y * 64;
    const int tid  = threadIdx.x;
    const int lane = tid & 31;
    const int w    = tid >> 5;
    const int wm   = w >> 1;            // 0..3
    const int wn   = w & 1;             // 0..1
    const int half = lane >> 4;
    const int l16  = lane & 15;

    v8f acc = {};

    for (int kk = 0; kk < H_; kk += 32) {
        __syncthreads();
        for (int idx = tid; idx < 64 * 32; idx += 256) {
            int r = idx >> 5, c = idx & 31;
            As[r][c] = prev_h[(size_t)(m0 + r) * H_ + kk + c];
        }
        for (int idx = tid; idx < 32 * 32; idx += 256) {
            int kr = idx >> 5, c = idx & 31;
            Bs[kr][c] = Wh[(size_t)(kk + kr) * H_ + h0 + c];
        }
        __syncthreads();

        const int ar = wm * 16 + l16;
        #pragma unroll
        for (int k4 = 0; k4 < 32; k4 += 4) {
            v2f a;      // A 16x4 f32 frag: lane M=l16, K = k4+2*half, +1
            a.x = As[ar][k4 + 2 * half];
            a.y = As[ar][k4 + 2 * half + 1];
            v2f bb;     // B 4x16 f32 frag: lane N=l16
            bb.x = Bs[k4 + 2 * half][wn * 16 + l16];
            bb.y = Bs[k4 + 2 * half + 1][wn * 16 + l16];
            acc = __builtin_amdgcn_wmma_f32_16x16x4_f32(
                      false, a, false, bb, (short)0, acc, false, false);
        }
    }

    const int h = h0 + wn * 16 + l16;
    const float bias = bh[h];
    #pragma unroll
    for (int v = 0; v < 8; ++v) {
        const int row = m0 + wm * 16 + v + (half ? 8 : 0);
        hp[(size_t)row * H_ + h] = acc[v] + bias;
    }
}

// ---------------------------------------------------------------------------
// Kernel A: fused Bahdanau attention for one batch element per workgroup.
//   - proj tile = batch_H[b] @ Wi                    (v_wmma_f32_16x16x32_bf16)
//   - e[t] = sum_h tanh(proj+hp)*Ws  (fused epilogue, shfl + LDS atomics)
//   - alpha = softmax(e)  -> written to d_out alpha region
//   - context = alpha @ batch_H[b]; xr = [context | onehot | prev_h] -> ws
// 512 threads = 16 waves: wave grid 4(M) x 4(N); macro tile 64(M) x 128(N).
// ---------------------------------------------------------------------------
__global__ __launch_bounds__(512)
void attn_kernel(const float* __restrict__ prev_h,
                 const float* __restrict__ batch_H,
                 const float* __restrict__ onehots,
                 const float* __restrict__ Wi,
                 const float* __restrict__ hp,     // [B,H] precomputed by hp_kernel
                 const float* __restrict__ Ws,
                 float* __restrict__ xr,           // [B, KEFF] workspace
                 float* __restrict__ out_alpha)    // [B, T]
{
    extern __shared__ unsigned char smem[];
    unsigned short* WiT = (unsigned short*)smem;      // [128][LDA] bf16, Wi chunk transposed (N-major)
    unsigned short* lA  = WiT + 128 * LDA;            // [ 64][LDA] bf16, batch_H[b] M-chunk
    float* hpS  = (float*)(lA + 64 * LDA);            // [H]
    float* wsS  = hpS + H_;                           // [H]
    float* eS   = wsS + H_;                           // [T] energies -> probs -> alpha
    float* redS = eS + T_;                            // [T] reduction scratch

    const int b    = blockIdx.x;
    const int tid  = threadIdx.x;
    const int lane = tid & 31;
    const int w    = tid >> 5;
    const int wm   = w >> 2;        // 0..3 : 16-row slice within 64-row chunk
    const int wn   = w & 3;         // 0..3 : 32-col slice within 128-col chunk
    const int half = lane >> 4;     // lane group (ISA A/B/C fragment layouts)
    const int l16  = lane & 15;

    // ---- Phase 0: load hp row + Ws, zero e ------------------------------
    hpS[tid] = hp[(size_t)b * H_ + tid];
    wsS[tid] = Ws[tid];
    if (tid < T_) eS[tid] = 0.0f;
    __syncthreads();

    const float* Ab = batch_H + (size_t)b * T_ * C_;

    // ---- Phase 1: fused proj GEMM + tanh-dot reduction ------------------
    for (int nc = 0; nc < 4; ++nc) {                       // 4 N-chunks of 128
        // stage Wi[:, nc*128 .. +128) transposed as bf16  (reads coalesced)
        for (int idx = tid; idx < 128 * C_; idx += 512) {
            int n = idx & 127, k = idx >> 7;
            WiT[n * LDA + k] = f2bf(Wi[k * H_ + nc * 128 + n]);
        }
        __syncthreads();

        for (int mc = 0; mc < 4; ++mc) {                   // 4 M-chunks of 64
            // stage batch_H[b, mc*64 .. +64, :] as bf16 (coalesced, L2-hot)
            for (int idx = tid; idx < 64 * C_; idx += 512) {
                int k = idx & 511, r = idx >> 9;
                lA[r * LDA + k] = f2bf(Ab[(size_t)(mc * 64 + r) * C_ + k]);
            }
            __syncthreads();

            v8f acc0 = {};      // 16x16 frag: cols n0..n0+15
            v8f acc1 = {};      // 16x16 frag: cols n0+16..n0+31
            const unsigned short* arow  = lA  + (wm * 16 + l16) * LDA;
            const unsigned short* b0row = WiT + (wn * 32 + l16) * LDA;
            const unsigned short* b1row = b0row + 16 * LDA;

            for (int kk = 0; kk < C_; kk += 32) {
                BFV a, b0, b1;
                // A 16x32 bf16 frag: lane holds M=l16, K = kk+8*half..+7 and kk+16+8*half..+7
                a.q[0]  = *(const uint4*)(arow  + kk + 8 * half);
                a.q[1]  = *(const uint4*)(arow  + kk + 16 + 8 * half);
                // B 32x16 bf16 frag: lane holds N=l16, K = kk+8*half..+7 and kk+16+8*half..+7 (WiT is N-major)
                b0.q[0] = *(const uint4*)(b0row + kk + 8 * half);
                b0.q[1] = *(const uint4*)(b0row + kk + 16 + 8 * half);
                b1.q[0] = *(const uint4*)(b1row + kk + 8 * half);
                b1.q[1] = *(const uint4*)(b1row + kk + 16 + 8 * half);
                acc0 = __builtin_amdgcn_wmma_f32_16x16x32_bf16(
                           false, a.v, false, b0.v, (short)0, acc0, false, false);
                acc1 = __builtin_amdgcn_wmma_f32_16x16x32_bf16(
                           false, a.v, false, b1.v, (short)0, acc1, false, false);
            }

            // epilogue: e[t] += sum over this wave's 32 columns of tanh(z+hp)*Ws
            // C/D layout: element v of frag -> row M = v + 8*half, col N = l16
            const int trow = mc * 64 + wm * 16 + (half ? 8 : 0);
            const int n0 = nc * 128 + wn * 32 + l16;
            const int n1 = n0 + 16;
            const float hp0 = hpS[n0], ws0 = wsS[n0];
            const float hp1 = hpS[n1], ws1 = wsS[n1];
            #pragma unroll
            for (int v = 0; v < 8; ++v) {
                float s = tanhf(acc0[v] + hp0) * ws0
                        + tanhf(acc1[v] + hp1) * ws1;
                // reduce across the 16 lanes of this half (same row, 16 cols)
                s += __shfl_xor(s, 1, 32);
                s += __shfl_xor(s, 2, 32);
                s += __shfl_xor(s, 4, 32);
                s += __shfl_xor(s, 8, 32);
                if (l16 == 0) atomicAdd(&eS[trow + v], s);   // ds_add_f32
            }
            __syncthreads();
        }
    }

    // ---- Phase 2: softmax over T=256 ------------------------------------
    if (tid < T_) redS[tid] = eS[tid];
    __syncthreads();
    for (int s = 128; s > 0; s >>= 1) {
        if (tid < s) redS[tid] = fmaxf(redS[tid], redS[tid + s]);
        __syncthreads();
    }
    const float maxv = redS[0];
    __syncthreads();
    float p = 0.0f;
    if (tid < T_) { p = expf(eS[tid] - maxv); redS[tid] = p; }
    __syncthreads();
    for (int s = 128; s > 0; s >>= 1) {
        if (tid < s) redS[tid] += redS[tid + s];
        __syncthreads();
    }
    const float inv_sum = 1.0f / redS[0];
    __syncthreads();
    if (tid < T_) {
        float al = p * inv_sum;
        eS[tid] = al;
        out_alpha[(size_t)b * T_ + tid] = al;
    }
    __syncthreads();

    // ---- Phase 3/4: context + pack xr = [context | onehot | prev_h] -----
    {
        float ctx = 0.0f;
        const float* col = Ab + tid;                 // coalesced across tid per t
        for (int t = 0; t < T_; ++t)
            ctx = fmaf(eS[t], col[(size_t)t * C_], ctx);
        float* xrow = xr + (size_t)b * KEFF;
        xrow[tid]       = ctx;                       // cols 0..511
        xrow[608 + tid] = prev_h[(size_t)b * H_ + tid];   // cols 608..1119
        if (tid < NC_) xrow[512 + tid] = onehots[(size_t)b * NC_ + tid]; // 512..607
    }
}

// ---------------------------------------------------------------------------
// Kernel B: LSTM step, z = xr @ [K;R] + bias, fused gates.
// Full fp32 via v_wmma_f32_16x16x4_f32 (outputs are the final h/c).
// Grid (H/32, B/64); 256 threads = 8 waves (4 M x 2 N). Each block computes
// its 32 h-columns for ALL FOUR gates so c/h fuse in registers.
// ---------------------------------------------------------------------------
__global__ __launch_bounds__(256)
void lstm_kernel(const float* __restrict__ xr,      // [B, KEFF]
                 const float* __restrict__ Km,      // [608, 2048]
                 const float* __restrict__ Rm,      // [512, 2048]
                 const float* __restrict__ bias,    // [2048]
                 const float* __restrict__ prev_c,  // [B, H]
                 float* __restrict__ out_h,         // [B, H]
                 float* __restrict__ out_c)         // [B, H]
{
    __shared__ float As[64][33];        // x rows, padded
    __shared__ float Bs[4][32][33];     // weight K-chunk for each gate

    const int h0 = blockIdx.x * 32;
    const int m0 = blockIdx.y * 64;
    const int tid  = threadIdx.x;
    const int lane = tid & 31;
    const int w    = tid >> 5;
    const int wm   = w >> 1;            // 0..3
    const int wn   = w & 1;             // 0..1
    const int half = lane >> 4;
    const int l16  = lane & 15;

    v8f acc[4] = {};                    // i, f, g, o accumulators (16x16 each)

    for (int kk = 0; kk < KEFF; kk += 32) {
        __syncthreads();
        for (int idx = tid; idx < 64 * 32; idx += 256) {
            int r = idx >> 5, c = idx & 31;
            As[r][c] = xr[(size_t)(m0 + r) * KEFF + kk + c];
        }
        for (int idx = tid; idx < 4 * 32 * 32; idx += 256) {
            int g = idx >> 10, rem = idx & 1023;
            int kr = rem >> 5, c = rem & 31;
            int k = kk + kr;
            int col = g * H_ + h0 + c;
            Bs[g][kr][c] = (k < 608) ? Km[(size_t)k * H4_ + col]
                                     : Rm[(size_t)(k - 608) * H4_ + col];
        }
        __syncthreads();

        const int ar = wm * 16 + l16;
        #pragma unroll
        for (int k4 = 0; k4 < 32; k4 += 4) {
            // A 16x4 f32 frag: lane M=l16, VGPR0/1 = K = k4+2*half, +1
            v2f a;
            a.x = As[ar][k4 + 2 * half];
            a.y = As[ar][k4 + 2 * half + 1];
            #pragma unroll
            for (int g = 0; g < 4; ++g) {
                v2f bb;   // B 4x16 f32 frag: lane N=l16, K = k4+2*half, +1
                bb.x = Bs[g][k4 + 2 * half][wn * 16 + l16];
                bb.y = Bs[g][k4 + 2 * half + 1][wn * 16 + l16];
                acc[g] = __builtin_amdgcn_wmma_f32_16x16x4_f32(
                             false, a, false, bb, (short)0, acc[g], false, false);
            }
        }
    }

    // fused gate epilogue (keras order i,f,g,o)
    const int h = h0 + wn * 16 + l16;
    const float bi = bias[0 * H_ + h];
    const float bf = bias[1 * H_ + h];
    const float bg = bias[2 * H_ + h];
    const float bo = bias[3 * H_ + h];
    #pragma unroll
    for (int v = 0; v < 8; ++v) {
        const int row = m0 + wm * 16 + v + (half ? 8 : 0);
        float zi = acc[0][v] + bi;
        float zf = acc[1][v] + bf;
        float zg = acc[2][v] + bg;
        float zo = acc[3][v] + bo;
        float cn = sigmoidf(zf) * prev_c[(size_t)row * H_ + h]
                 + sigmoidf(zi) * tanhf(zg);
        float hn = sigmoidf(zo) * tanhf(cn);
        out_h[(size_t)row * H_ + h] = hn;
        out_c[(size_t)row * H_ + h] = cn;
    }
}

// ---------------------------------------------------------------------------
extern "C" void kernel_launch(void* const* d_in, const int* in_sizes, int n_in,
                              void* d_out, int out_size, void* d_ws, size_t ws_size,
                              hipStream_t stream) {
    (void)in_sizes; (void)n_in; (void)out_size; (void)ws_size;
    const float* prev_h  = (const float*)d_in[0];
    const float* prev_c  = (const float*)d_in[1];
    const float* batch_H = (const float*)d_in[2];
    const float* onehots = (const float*)d_in[3];
    const float* Wi      = (const float*)d_in[4];
    const float* Wh      = (const float*)d_in[5];
    const float* bh      = (const float*)d_in[6];
    const float* Ws      = (const float*)d_in[7];
    const float* Km      = (const float*)d_in[8];
    const float* Rm      = (const float*)d_in[9];
    const float* bias    = (const float*)d_in[10];

    float* out       = (float*)d_out;
    float* out_h     = out;                    // [B,H]
    float* out_c     = out + B_ * H_;          // [B,H]
    float* out_alpha = out + 2 * B_ * H_;      // [B,T]
    float* xr        = (float*)d_ws;           // [B, KEFF]
    float* hp        = xr + (size_t)B_ * KEFF; // [B, H]

    const size_t smemA = (size_t)(128 + 64) * LDA * sizeof(unsigned short)
                       + (size_t)(H_ + H_ + T_ + T_) * sizeof(float);  // ~209 KB of 320 KB

    hp_kernel<<<dim3(H_ / 32, B_ / 64), dim3(256), 0, stream>>>(
        prev_h, Wh, bh, hp);
    attn_kernel<<<dim3(B_), dim3(512), smemA, stream>>>(
        prev_h, batch_H, onehots, Wi, hp, Ws, xr, out_alpha);
    lstm_kernel<<<dim3(H_ / 32, B_ / 64), dim3(256), 0, stream>>>(
        xr, Km, Rm, bias, prev_c, out_h, out_c);
}
